// CLKG_compgatv3_DFconv3D_72404558676348
// MI455X (gfx1250) — matrix operand order repaired and checked by reading, not compiled, not run on hardware
//
#include <hip/hip_runtime.h>

// ---------------------------------------------------------------------------
// Problem constants (match reference)
// ---------------------------------------------------------------------------
#define ENTN   50000
#define RELN   500
#define DIMK   200
#define NEDGE  200000
#define BATCH  2048
#define FCCH   32
#define OD     4
#define OH     8
#define OW     18
#define NFEAT  (FCCH * OD * OH * OW)   // 18432
#define BNEPS  1e-5f

// ---------------------------------------------------------------------------
// WMMA types (CDNA5 / gfx1250, wave32)
// ---------------------------------------------------------------------------
typedef __attribute__((ext_vector_type(16))) _Float16 v16h;
typedef __attribute__((ext_vector_type(8)))  _Float16 v8h;
typedef __attribute__((ext_vector_type(8)))  float    v8f;
typedef int v4i __attribute__((vector_size(16)));   // matches async-LDS builtin param

#define TM 64
#define TN 64
#define TK 32
#define APITCH (TK + 8)          // halves; 80B rows keep 16B alignment, spread banks
#define BHALFBLK (TN * 16 + 8)   // halves per k-half block of sB (+16B pad)

// Async global->LDS staging (CDNA5 GLOBAL_LOAD_ASYNC_TO_LDS_*, ASYNCcnt)
#if __has_builtin(__builtin_amdgcn_global_load_async_to_lds_b128)
#define HAVE_ASYNC_LDS 1
#endif
#define AS1 __attribute__((address_space(1)))
#define AS3 __attribute__((address_space(3)))
#if __has_builtin(__builtin_amdgcn_s_wait_asynccnt)
#define ASYNC_WAIT() __builtin_amdgcn_s_wait_asynccnt(0)
#else
#define ASYNC_WAIT() asm volatile("s_wait_asynccnt 0x0" ::: "memory")
#endif

__device__ __forceinline__ void cvt16(const float4& x0, const float4& x1,
                                      const float4& x2, const float4& x3,
                                      v8h& lo, v8h& hi)
{
    lo[0] = (_Float16)x0.x; lo[1] = (_Float16)x0.y; lo[2] = (_Float16)x0.z; lo[3] = (_Float16)x0.w;
    lo[4] = (_Float16)x1.x; lo[5] = (_Float16)x1.y; lo[6] = (_Float16)x1.z; lo[7] = (_Float16)x1.w;
    hi[0] = (_Float16)x2.x; hi[1] = (_Float16)x2.y; hi[2] = (_Float16)x2.z; hi[3] = (_Float16)x2.w;
    hi[4] = (_Float16)x3.x; hi[5] = (_Float16)x3.y; hi[6] = (_Float16)x3.z; hi[7] = (_Float16)x3.w;
}

// ---------------------------------------------------------------------------
// Generic WMMA GEMM:  C[M,N] = A[M,K] @ B  (+ optional per-N bias)
//   BT=false: B is K x N row-major (ldb = row stride)
//   BT=true : B is N x K row-major (computes A @ B^T)
//   GATHER=true: A row e = gent[gsrc[e]] - grel[gtype[e]]  (stride lda)
// Block: 128 threads = 4 waves; each wave computes a 16x64 strip of a
// 64x64 tile via 4x v_wmma_f32_16x16x32_f16 per K-chunk.
// sB is fragment-major so each B fragment is 2x ds_load_b128; tile loaders
// take an unconditional float4 fast path on interior tiles (one wait),
// and interior A tiles stream through GLOBAL_LOAD_ASYNC_TO_LDS_B128.
// ---------------------------------------------------------------------------
template <bool BT, bool GATHER>
__global__ __launch_bounds__(128)
void wmma_gemm_kernel(const float* __restrict__ A,
                      const float* __restrict__ B,
                      float* __restrict__ C,
                      const float* __restrict__ bias,
                      int M, int N, int K,
                      int lda, int ldb, int ldc,
                      const int* __restrict__ gsrc,
                      const int* __restrict__ gtype,
                      const float* __restrict__ gent,
                      const float* __restrict__ grel)
{
    __shared__ __align__(16) _Float16 sA[TM * APITCH];
    __shared__ __align__(16) _Float16 sB[2 * BHALFBLK];
#ifdef HAVE_ASYNC_LDS
    __shared__ __align__(16) float sAraw[TM * TK];     // raw f32 staging for async path
#endif

    const int tid     = threadIdx.x;
    const int wave    = tid >> 5;
    const int lane    = tid & 31;
    const int half_id = lane >> 4;
    const int r       = lane & 15;
    const int m_base  = blockIdx.y * TM;
    const int n_base  = blockIdx.x * TN;

    v8f acc[4] = {{}, {}, {}, {}};

    for (int kc = 0; kc < K; kc += TK) {
        // ================= A tile: 64 rows x 32 k, f32 -> f16 ==============
#ifdef HAVE_ASYNC_LDS
        const bool fullA = (!GATHER) && (m_base + TM <= M) && (kc + TK <= K);
        if (fullA) {
            // 512 b128 transfers: async copy raw f32 tile into LDS staging
#pragma unroll
            for (int it = 0; it < 4; ++it) {
                int c   = tid + it * 128;
                int row = c >> 3, ci = c & 7;
                const float* gp = A + (size_t)(m_base + row) * lda + kc + ci * 4;
                float*       lp = &sAraw[row * TK + ci * 4];
                __builtin_amdgcn_global_load_async_to_lds_b128(
                    (AS1 v4i*)(v4i*)(float*)gp, (AS3 v4i*)(v4i*)lp, 0, 0);
            }
            ASYNC_WAIT();
            __syncthreads();
            // repack f32 staging -> f16 fragment rows (vector LDS stores)
            int row  = tid >> 1;
            int col0 = (tid & 1) * 16;
            const float* sp = &sAraw[row * TK + col0];
            v8h lo, hi;
#pragma unroll
            for (int i = 0; i < 8; ++i) { lo[i] = (_Float16)sp[i]; hi[i] = (_Float16)sp[8 + i]; }
            _Float16* ap16 = &sA[row * APITCH + col0];
            *(v8h*)ap16 = lo;
            *(v8h*)(ap16 + 8) = hi;
        } else
#endif
        {
            int row  = tid >> 1;
            int col0 = (tid & 1) * 16;
            int gm   = m_base + row;
            const bool fullK = (kc + col0 + 16 <= K);
            v8h lo = {}, hi = {};
            if (!GATHER) {
                const float* ap = A + (size_t)gm * lda + kc + col0;
                if (gm < M && fullK) {
                    // interior: 4 unconditional b128 loads, single wait
                    __builtin_prefetch(ap + TK, 0, 3);
                    const float4* ap4 = (const float4*)ap;
                    float4 x0 = ap4[0], x1 = ap4[1], x2 = ap4[2], x3 = ap4[3];
                    cvt16(x0, x1, x2, x3, lo, hi);
                } else if (gm < M) {
#pragma unroll
                    for (int i = 0; i < 16; ++i) {
                        int k = kc + col0 + i;
                        float v = (k < K) ? ap[i] : 0.0f;
                        if (i < 8) lo[i] = (_Float16)v; else hi[i - 8] = (_Float16)v;
                    }
                }
            } else {
                if (gm < M) {
                    int se = gsrc[gm], te = gtype[gm];
                    const float* ep = gent + (size_t)se * lda + kc + col0;
                    const float* rp = grel + (size_t)te * lda + kc + col0;
                    if (fullK) {
                        const float4* ep4 = (const float4*)ep;
                        const float4* rp4 = (const float4*)rp;
                        float4 e0 = ep4[0], e1 = ep4[1], e2 = ep4[2], e3 = ep4[3];
                        float4 r0 = rp4[0], r1 = rp4[1], r2 = rp4[2], r3 = rp4[3];
                        float4 d0, d1, d2, d3;
                        d0.x = e0.x - r0.x; d0.y = e0.y - r0.y; d0.z = e0.z - r0.z; d0.w = e0.w - r0.w;
                        d1.x = e1.x - r1.x; d1.y = e1.y - r1.y; d1.z = e1.z - r1.z; d1.w = e1.w - r1.w;
                        d2.x = e2.x - r2.x; d2.y = e2.y - r2.y; d2.z = e2.z - r2.z; d2.w = e2.w - r2.w;
                        d3.x = e3.x - r3.x; d3.y = e3.y - r3.y; d3.z = e3.z - r3.z; d3.w = e3.w - r3.w;
                        cvt16(d0, d1, d2, d3, lo, hi);
                    } else {
#pragma unroll
                        for (int i = 0; i < 16; ++i) {
                            int k = kc + col0 + i;
                            float v = (k < K) ? (ep[i] - rp[i]) : 0.0f;
                            if (i < 8) lo[i] = (_Float16)v; else hi[i - 8] = (_Float16)v;
                        }
                    }
                }
            }
            _Float16* ap16 = &sA[row * APITCH + col0];
            *(v8h*)ap16 = lo;
            *(v8h*)(ap16 + 8) = hi;
        }

        // ========== B tile -> fragment-major f16 layout in LDS =============
        {
            v8h lo = {}, hi = {};
            int n_local, halfb;
            if (!BT) {
                // B is K x N: each thread gathers one 16-deep column strip
                // (coalesced across lanes at each step)
                n_local = tid & 63;
                halfb   = tid >> 6;               // which k-half (0 or 1)
                int gn  = n_base + n_local;
                int kb  = kc + halfb * 16;
                const float* bp = B + (size_t)kb * ldb + gn;
                if (gn < N && kb + 16 <= K) {
                    float t[16];
#pragma unroll
                    for (int i = 0; i < 16; ++i) t[i] = bp[(size_t)i * ldb];  // batched, one wait
#pragma unroll
                    for (int i = 0; i < 8; ++i) { lo[i] = (_Float16)t[i]; hi[i] = (_Float16)t[8 + i]; }
                } else if (gn < N) {
#pragma unroll
                    for (int i = 0; i < 16; ++i) {
                        float v = (kb + i < K) ? bp[(size_t)i * ldb] : 0.0f;
                        if (i < 8) lo[i] = (_Float16)v; else hi[i - 8] = (_Float16)v;
                    }
                }
            } else {
                // B is N x K: 16 consecutive floats per thread
                n_local = tid >> 1;
                halfb   = tid & 1;
                int gn  = n_base + n_local;
                int kb  = kc + halfb * 16;
                const float* bp = B + (size_t)gn * ldb + kb;
                if (gn < N && kb + 16 <= K) {
                    __builtin_prefetch(bp + TK, 0, 3);
                    const float4* bp4 = (const float4*)bp;
                    float4 x0 = bp4[0], x1 = bp4[1], x2 = bp4[2], x3 = bp4[3];
                    cvt16(x0, x1, x2, x3, lo, hi);
                } else if (gn < N) {
#pragma unroll
                    for (int i = 0; i < 16; ++i) {
                        float v = (kb + i < K) ? bp[i] : 0.0f;
                        if (i < 8) lo[i] = (_Float16)v; else hi[i - 8] = (_Float16)v;
                    }
                }
            }
            _Float16* cb = &sB[halfb * BHALFBLK + n_local * 16];
            *(v8h*)cb = lo;
            *(v8h*)(cb + 8) = hi;
        }
        __syncthreads();

        // ---- A fragment (ISA 7.12.2: 16-bit A 16x32, wave32) ----
        v16h af;
        {
            const _Float16* arow = &sA[(wave * 16 + r) * APITCH];
            int b0 = half_id * 16;
#pragma unroll
            for (int i = 0; i < 8; ++i) af[i]     = arow[b0 + i];
#pragma unroll
            for (int i = 0; i < 8; ++i) af[8 + i] = arow[b0 + 8 + i];
        }
        // ---- load all 4 B fragments (8x ds_load_b128, clauseable) ----
        v16h bfr[4];
#pragma unroll
        for (int nb = 0; nb < 4; ++nb) {
            const _Float16* cb = &sB[half_id * BHALFBLK + (nb * 16 + r) * 16];
            v8h lo = *(const v8h*)cb;
            v8h hi = *(const v8h*)(cb + 8);
            bfr[nb] = __builtin_shufflevector(lo, hi,
                0, 1, 2, 3, 4, 5, 6, 7, 8, 9, 10, 11, 12, 13, 14, 15);
        }
        // ---- 4x WMMA (independent accumulator chains) ----
#pragma unroll
        for (int nb = 0; nb < 4; ++nb) {
            acc[nb] = __builtin_amdgcn_wmma_f32_16x16x32_f16(
                false, af, false, bfr[nb], (short)0, acc[nb], false, false);
        }
        __syncthreads();
    }

    // ---- epilogue (C/D layout: VGPR j -> M = half_id*8 + j, N = r) ----
#pragma unroll
    for (int nb = 0; nb < 4; ++nb) {
        int gn = n_base + nb * 16 + r;
        if (gn >= N) continue;
        float bv = bias ? bias[gn] : 0.0f;
#pragma unroll
        for (int j = 0; j < 8; ++j) {
            int gm = m_base + wave * 16 + half_id * 8 + j;
            if (gm < M) C[(size_t)gm * ldc + gn] = acc[nb][j] + bv;
        }
    }
}

// ---------------------------------------------------------------------------
// Utility / pipeline kernels
// ---------------------------------------------------------------------------
__global__ void fill_kernel(float* p, float v, int n) {
    int i = blockIdx.x * blockDim.x + threadIdx.x;
    if (i < n) p[i] = v;
}

// score[e] = leaky_relu( sum_n tanh(m[e,n] + entW[dst[e],n]) * att[n] , 0.2)
__global__ __launch_bounds__(256)
void edge_score_kernel(const float* __restrict__ m, const float* __restrict__ entW,
                       const float* __restrict__ att, const int* __restrict__ dst,
                       float* __restrict__ score, int E, int dim)
{
    int widx = (blockIdx.x * blockDim.x + threadIdx.x) >> 5;
    int lane = threadIdx.x & 31;
    if (widx >= E) return;
    int d = dst[widx];
    const float* mrow = m + (size_t)widx * dim;
    const float* hrow = entW + (size_t)d * dim;
    float s = 0.0f;
    for (int n = lane; n < dim; n += 32)
        s += tanhf(mrow[n] + hrow[n]) * att[n];
#pragma unroll
    for (int off = 16; off > 0; off >>= 1) s += __shfl_xor(s, off, 32);
    if (lane == 0) score[widx] = (s >= 0.0f) ? s : 0.2f * s;
}

__global__ void seg_max_kernel(const float* __restrict__ score, const int* __restrict__ dst,
                               float* __restrict__ smax, int E)
{
    int e = blockIdx.x * blockDim.x + threadIdx.x;
    if (e >= E) return;
    __hip_atomic_fetch_max(&smax[dst[e]], score[e],
                           __ATOMIC_RELAXED, __HIP_MEMORY_SCOPE_AGENT);
}

__global__ void seg_exp_kernel(float* __restrict__ score, const int* __restrict__ dst,
                               const float* __restrict__ smax, float* __restrict__ denom, int E)
{
    int e = blockIdx.x * blockDim.x + threadIdx.x;
    if (e >= E) return;
    int d = dst[e];
    float ex = __expf(score[e] - smax[d]);
    score[e] = ex;
    atomicAdd(&denom[d], ex);
}

// agg[dst[e],:] += alpha_e * m[e,:]
__global__ __launch_bounds__(256)
void seg_agg_kernel(const float* __restrict__ m, const float* __restrict__ ex,
                    const float* __restrict__ denom, const int* __restrict__ dst,
                    float* __restrict__ agg, int E, int dim)
{
    int e = blockIdx.x;
    if (e >= E) return;
    int d = dst[e];
    float alpha = ex[e] / (denom[d] + 1e-16f);
    const float* mrow = m + (size_t)e * dim;
    float* arow = agg + (size_t)d * dim;
    for (int n = threadIdx.x; n < dim; n += blockDim.x)
        atomicAdd(&arow[n], alpha * mrow[n]);
}

__global__ void ent_out_kernel(const float* __restrict__ agg, const float* __restrict__ bias,
                               float* __restrict__ ent, int total, int dim)
{
    int i = blockIdx.x * blockDim.x + threadIdx.x;
    if (i < total) ent[i] = tanhf(agg[i] + bias[i % dim]);
}

__global__ void gather_kernel(const float* __restrict__ ent, const float* __restrict__ rel,
                              const int* __restrict__ h, const int* __restrict__ rr,
                              const int* __restrict__ t,
                              float* __restrict__ head, float* __restrict__ tail,
                              float* __restrict__ rvec, int B, int dim)
{
    int i = blockIdx.x * blockDim.x + threadIdx.x;
    if (i >= B * dim) return;
    int b = i / dim, c = i % dim;
    head[i] = ent[(size_t)h[b] * dim + c];
    tail[i] = ent[(size_t)t[b] * dim + c];
    rvec[i] = rel[(size_t)rr[b] * dim + c];
}

__global__ void rowsq_kernel(const float* __restrict__ X, float* __restrict__ out,
                             int rows, int cols)
{
    int row  = (blockIdx.x * blockDim.x + threadIdx.x) >> 5;
    int lane = threadIdx.x & 31;
    if (row >= rows) return;
    float s = 0.0f;
    for (int c = lane; c < cols; c += 32) { float v = X[(size_t)row * cols + c]; s += v * v; }
#pragma unroll
    for (int off = 16; off > 0; off >>= 1) s += __shfl_xor(s, off, 32);
    if (lane == 0) out[row] = s;
}

// out = exp(-0.5 * max(||x||^2 + ||c||^2 - 2*dot, 0) / sigma^2)
__global__ void gauss_kernel(const float* __restrict__ dots, const float* __restrict__ xsq,
                             const float* __restrict__ csq, const float* __restrict__ sig,
                             float* __restrict__ out, int B, int nc)
{
    int i = blockIdx.x * blockDim.x + threadIdx.x;
    if (i >= B * nc) return;
    int b = i / nc, n = i % nc;
    float sq = fmaxf(xsq[b] + csq[n] - 2.0f * dots[i], 0.0f);
    float s  = sig[n];
    out[i] = __expf(-0.5f * sq / (s * s));
}

// stack channels: [rvec, head, r_g, head_g, rvec]
__global__ void stack_kernel(const float* __restrict__ rvec, const float* __restrict__ head,
                             const float* __restrict__ rg, const float* __restrict__ hg,
                             float* __restrict__ xs, int B, int dim)
{
    int i = blockIdx.x * blockDim.x + threadIdx.x;
    int total = B * 5 * dim;
    if (i >= total) return;
    int b = i / (5 * dim);
    int rem = i % (5 * dim);
    int dch = rem / dim;
    int c = rem % dim;
    const float* src = (dch == 1) ? head : (dch == 2) ? rg : (dch == 3) ? hg : rvec;
    xs[i] = src[(size_t)b * dim + c];
}

__global__ __launch_bounds__(256)
void gstats_kernel(const float* __restrict__ x, float* __restrict__ sums, int n)
{
    __shared__ float ss[256], sq[256];
    float a = 0.0f, q = 0.0f;
    for (int i = blockIdx.x * blockDim.x + threadIdx.x; i < n; i += gridDim.x * blockDim.x) {
        float v = x[i]; a += v; q += v * v;
    }
    ss[threadIdx.x] = a; sq[threadIdx.x] = q;
    __syncthreads();
    for (int s = 128; s > 0; s >>= 1) {
        if (threadIdx.x < s) { ss[threadIdx.x] += ss[threadIdx.x + s]; sq[threadIdx.x] += sq[threadIdx.x + s]; }
        __syncthreads();
    }
    if (threadIdx.x == 0) { atomicAdd(&sums[0], ss[0]); atomicAdd(&sums[1], sq[0]); }
}

__global__ void gnorm_kernel(float* __restrict__ x, const float* __restrict__ sums, int n)
{
    int i = blockIdx.x * blockDim.x + threadIdx.x;
    if (i >= n) return;
    float mu  = sums[0] / (float)n;
    float var = sums[1] / (float)n - mu * mu;
    x[i] = (x[i] - mu) * rsqrtf(var + BNEPS);
}

// direct conv3d: in (B,1,5,10,20), w (32,1,2,3,3), out (B,32,4,8,18)
__global__ __launch_bounds__(256)
void conv3d_kernel(const float* __restrict__ xs, const float* __restrict__ w,
                   float* __restrict__ out, int B)
{
    __shared__ float sw[FCCH * 18];
    for (int i = threadIdx.x; i < FCCH * 18; i += blockDim.x) sw[i] = w[i];
    __syncthreads();
    int idx = blockIdx.x * blockDim.x + threadIdx.x;
    int total = B * FCCH * OD * OH * OW;
    if (idx >= total) return;
    int j = idx % OW; int tmp = idx / OW;
    int i = tmp % OH; tmp /= OH;
    int d = tmp % OD; tmp /= OD;
    int c = tmp % FCCH; int b = tmp / FCCH;
    const float* xb = xs + (size_t)b * 5 * 10 * 20;
    const float* wc = sw + c * 18;
    float acc = 0.0f;
#pragma unroll
    for (int dd = 0; dd < 2; ++dd)
#pragma unroll
        for (int di = 0; di < 3; ++di)
#pragma unroll
            for (int dj = 0; dj < 3; ++dj)
                acc += xb[(d + dd) * 200 + (i + di) * 20 + (j + dj)] * wc[dd * 9 + di * 3 + dj];
    out[idx] = acc;
}

// per-channel stats: one block per (b,c) slab of 576 elems
__global__ __launch_bounds__(256)
void cstats_kernel(const float* __restrict__ x, float* __restrict__ st, int B)
{
    int bc = blockIdx.x;
    int c  = bc % FCCH;
    const float* p = x + (size_t)bc * (OD * OH * OW);
    float a = 0.0f, q = 0.0f;
    for (int i = threadIdx.x; i < OD * OH * OW; i += blockDim.x) { float v = p[i]; a += v; q += v * v; }
    __shared__ float ss[256], sq[256];
    ss[threadIdx.x] = a; sq[threadIdx.x] = q;
    __syncthreads();
    for (int s = 128; s > 0; s >>= 1) {
        if (threadIdx.x < s) { ss[threadIdx.x] += ss[threadIdx.x + s]; sq[threadIdx.x] += sq[threadIdx.x + s]; }
        __syncthreads();
    }
    if (threadIdx.x == 0) { atomicAdd(&st[c], ss[0]); atomicAdd(&st[FCCH + c], sq[0]); }
}

__global__ void cbn_kernel(float* __restrict__ x, const float* __restrict__ st,
                           const float* __restrict__ relu_a, int B)
{
    int idx = blockIdx.x * blockDim.x + threadIdx.x;
    int total = B * FCCH * OD * OH * OW;
    if (idx >= total) return;
    int c = (idx / (OD * OH * OW)) % FCCH;
    float cnt = (float)B * (float)(OD * OH * OW);
    float mu  = st[c] / cnt;
    float var = st[FCCH + c] / cnt - mu * mu;
    float v = (x[idx] - mu) * rsqrtf(var + BNEPS);
    float a = relu_a[0];
    x[idx] = (v >= 0.0f) ? v : a * v;
}

// fused per-column BN -> PReLU -> BN over the batch axis (B == 2048 == 256*8)
__global__ __launch_bounds__(256)
void colbn_kernel(const float* __restrict__ y, float* __restrict__ out,
                  const float* __restrict__ prelu_a, int B, int dim)
{
    int n   = blockIdx.x;
    int tid = threadIdx.x;
    const int RPT = 8;
    float v[RPT];
    float a = 0.0f, q = 0.0f;
    __shared__ float ss[256], sq[256];
#pragma unroll
    for (int k = 0; k < RPT; ++k) {
        v[k] = y[(size_t)(tid + k * 256) * dim + n];
        a += v[k]; q += v[k] * v[k];
    }
    ss[tid] = a; sq[tid] = q;
    __syncthreads();
    for (int s = 128; s > 0; s >>= 1) {
        if (tid < s) { ss[tid] += ss[tid + s]; sq[tid] += sq[tid + s]; }
        __syncthreads();
    }
    float mu = ss[0] / (float)B, var = sq[0] / (float)B - (ss[0] / (float)B) * (ss[0] / (float)B);
    __syncthreads();
    float pa = prelu_a[0];
    float rstd = rsqrtf(var + BNEPS);
    a = 0.0f; q = 0.0f;
#pragma unroll
    for (int k = 0; k < RPT; ++k) {
        float z = (v[k] - mu) * rstd;
        z = (z >= 0.0f) ? z : pa * z;
        v[k] = z; a += z; q += z * z;
    }
    ss[tid] = a; sq[tid] = q;
    __syncthreads();
    for (int s = 128; s > 0; s >>= 1) {
        if (tid < s) { ss[tid] += ss[tid + s]; sq[tid] += sq[tid + s]; }
        __syncthreads();
    }
    float mu2 = ss[0] / (float)B, var2 = sq[0] / (float)B - (ss[0] / (float)B) * (ss[0] / (float)B);
    float rstd2 = rsqrtf(var2 + BNEPS);
#pragma unroll
    for (int k = 0; k < RPT; ++k)
        out[(size_t)(tid + k * 256) * dim + n] = (v[k] - mu2) * rstd2;
}

// ---------------------------------------------------------------------------
// Host launcher
// ---------------------------------------------------------------------------
static inline int cdiv(int a, int b) { return (a + b - 1) / b; }

extern "C" void kernel_launch(void* const* d_in, const int* in_sizes, int n_in,
                              void* d_out, int out_size, void* d_ws, size_t ws_size,
                              hipStream_t stream)
{
    const float* ent_emb = (const float*)d_in[0];
    const float* rel_emb = (const float*)d_in[1];
    const float* W1      = (const float*)d_in[2];
    const float* Wrel1   = (const float*)d_in[3];
    const float* att1    = (const float*)d_in[4];
    const float* bias1   = (const float*)d_in[5];
    const float* ent_gc  = (const float*)d_in[6];
    const float* ent_gs  = (const float*)d_in[7];
    const float* rel_gc  = (const float*)d_in[8];
    const float* rel_gs  = (const float*)d_in[9];
    const float* conv_w  = (const float*)d_in[10];
    const float* proj_w  = (const float*)d_in[11];
    const float* proj_b  = (const float*)d_in[12];
    const float* prelu_a = (const float*)d_in[13];
    const float* relu_a  = (const float*)d_in[14];
    const float* b_out   = (const float*)d_in[15];
    const int*   e_src   = (const int*)d_in[16];            // edge_index[0]
    const int*   e_dst   = (const int*)d_in[16] + NEDGE;    // edge_index[1]
    const int*   e_type  = (const int*)d_in[17];
    const int*   h_idx   = (const int*)d_in[18];
    const int*   r_idx   = (const int*)d_in[19];
    const int*   t_idx   = (const int*)d_in[20];

    // output layout: cl_x | scores | tail | head | ent | rel
    float* out      = (float*)d_out;
    float* o_clx    = out;
    float* o_scores = out + (size_t)BATCH * DIMK;
    float* o_tail   = o_scores + (size_t)BATCH * ENTN;
    float* o_head   = o_tail   + (size_t)BATCH * DIMK;
    float* o_ent    = o_head   + (size_t)BATCH * DIMK;
    float* o_rel    = o_ent    + (size_t)ENTN * DIMK;

    // workspace layout (floats); mbuf is reused for conv features,
    // entW is reused as the aggregation buffer, dots reused as proj output.
    float* w = (float*)d_ws;
    size_t off = 0;
    float* entW   = w + off; off += (size_t)ENTN * DIMK;      // 10M (also: agg)
    float* mbuf   = w + off; off += (size_t)NEDGE * DIMK;     // 40M (also: conv feats)
    float* score  = w + off; off += NEDGE;
    float* smax   = w + off; off += ENTN;
    float* denom  = w + off; off += ENTN;
    float* rvec   = w + off; off += (size_t)BATCH * DIMK;
    float* hh     = w + off; off += BATCH;
    float* rr2    = w + off; off += BATCH;
    float* ccE    = w + off; off += 256;
    float* ccR    = w + off; off += 256;
    float* dots   = w + off; off += (size_t)BATCH * DIMK;     // also: proj output y
    float* hg     = w + off; off += (size_t)BATCH * DIMK;
    float* rg     = w + off; off += (size_t)BATCH * DIMK;
    float* xstack = w + off; off += (size_t)BATCH * 5 * DIMK;
    float* bn1s   = w + off; off += 8;
    float* chst   = w + off; off += 2 * FCCH;
    float* agg    = entW;
    float* convb  = mbuf;
    float* ybuf   = dots;

    // ---- init scratch that must be zero / -inf ----
    fill_kernel<<<cdiv(ENTN, 256), 256, 0, stream>>>(smax, -1e30f, ENTN);
    fill_kernel<<<cdiv(ENTN, 256), 256, 0, stream>>>(denom, 0.0f, ENTN);
    fill_kernel<<<1, 32, 0, stream>>>(bn1s, 0.0f, 8);
    fill_kernel<<<1, 64, 0, stream>>>(chst, 0.0f, 2 * FCCH);

    // ---- 1) entW = ent_emb @ W1 ----
    {
        dim3 g(cdiv(DIMK, TN), cdiv(ENTN, TM));
        wmma_gemm_kernel<false, false><<<g, 128, 0, stream>>>(
            ent_emb, W1, entW, nullptr, ENTN, DIMK, DIMK, DIMK, DIMK, DIMK,
            nullptr, nullptr, nullptr, nullptr);
    }
    // ---- 2) rel' = rel_emb @ Wrel1 -> output ----
    {
        dim3 g(cdiv(DIMK, TN), cdiv(RELN, TM));
        wmma_gemm_kernel<false, false><<<g, 128, 0, stream>>>(
            rel_emb, Wrel1, o_rel, nullptr, RELN, DIMK, DIMK, DIMK, DIMK, DIMK,
            nullptr, nullptr, nullptr, nullptr);
    }
    // ---- 3) m = (ent[src] - rel[etype]) @ W1  (gathered-A GEMM) ----
    {
        dim3 g(cdiv(DIMK, TN), cdiv(NEDGE, TM));
        wmma_gemm_kernel<false, true><<<g, 128, 0, stream>>>(
            nullptr, W1, mbuf, nullptr, NEDGE, DIMK, DIMK, DIMK, DIMK, DIMK,
            e_src, e_type, ent_emb, rel_emb);
    }
    // ---- 4) attention scores + segment softmax ----
    edge_score_kernel<<<cdiv(NEDGE, 8), 256, 0, stream>>>(mbuf, entW, att1, e_dst, score, NEDGE, DIMK);
    fill_kernel<<<cdiv(ENTN * DIMK, 256), 256, 0, stream>>>(agg, 0.0f, ENTN * DIMK); // entW dead now
    seg_max_kernel<<<cdiv(NEDGE, 256), 256, 0, stream>>>(score, e_dst, smax, NEDGE);
    seg_exp_kernel<<<cdiv(NEDGE, 256), 256, 0, stream>>>(score, e_dst, smax, denom, NEDGE);
    seg_agg_kernel<<<NEDGE, 256, 0, stream>>>(mbuf, score, denom, e_dst, agg, NEDGE, DIMK);
    ent_out_kernel<<<cdiv(ENTN * DIMK, 256), 256, 0, stream>>>(agg, bias1, o_ent, ENTN * DIMK, DIMK);

    // ---- 5) gather head / tail / rvec ----
    gather_kernel<<<cdiv(BATCH * DIMK, 256), 256, 0, stream>>>(
        o_ent, o_rel, h_idx, r_idx, t_idx, o_head, o_tail, rvec, BATCH, DIMK);

    // ---- 6) gaussian features (||.||^2 via rowsq + WMMA dot GEMMs) ----
    rowsq_kernel<<<cdiv(BATCH * 32, 256), 256, 0, stream>>>(o_head, hh, BATCH, DIMK);
    rowsq_kernel<<<cdiv(BATCH * 32, 256), 256, 0, stream>>>(rvec, rr2, BATCH, DIMK);
    rowsq_kernel<<<cdiv(DIMK * 32, 256), 256, 0, stream>>>(ent_gc, ccE, DIMK, DIMK);
    rowsq_kernel<<<cdiv(DIMK * 32, 256), 256, 0, stream>>>(rel_gc, ccR, DIMK, DIMK);
    {
        dim3 g(cdiv(DIMK, TN), cdiv(BATCH, TM));
        wmma_gemm_kernel<true, false><<<g, 128, 0, stream>>>(
            o_head, ent_gc, dots, nullptr, BATCH, DIMK, DIMK, DIMK, DIMK, DIMK,
            nullptr, nullptr, nullptr, nullptr);
        gauss_kernel<<<cdiv(BATCH * DIMK, 256), 256, 0, stream>>>(dots, hh, ccE, ent_gs, hg, BATCH, DIMK);
        wmma_gemm_kernel<true, false><<<g, 128, 0, stream>>>(
            rvec, rel_gc, dots, nullptr, BATCH, DIMK, DIMK, DIMK, DIMK, DIMK,
            nullptr, nullptr, nullptr, nullptr);
        gauss_kernel<<<cdiv(BATCH * DIMK, 256), 256, 0, stream>>>(dots, rr2, ccR, rel_gs, rg, BATCH, DIMK);
    }

    // ---- 7) stack + BN(all) ----
    int nstack = BATCH * 5 * DIMK;
    stack_kernel<<<cdiv(nstack, 256), 256, 0, stream>>>(rvec, o_head, rg, hg, xstack, BATCH, DIMK);
    gstats_kernel<<<1024, 256, 0, stream>>>(xstack, bn1s, nstack);
    gnorm_kernel<<<cdiv(nstack, 256), 256, 0, stream>>>(xstack, bn1s, nstack);

    // ---- 8) conv3d + BN(per-channel) + PReLU ----
    int nconv = BATCH * NFEAT;
    conv3d_kernel<<<cdiv(nconv, 256), 256, 0, stream>>>(xstack, conv_w, convb, BATCH);
    cstats_kernel<<<BATCH * FCCH, 256, 0, stream>>>(convb, chst, BATCH);
    cbn_kernel<<<cdiv(nconv, 256), 256, 0, stream>>>(convb, chst, relu_a, BATCH);

    // ---- 9) projection GEMM (K = 18432) + bias ----
    {
        dim3 g(cdiv(DIMK, TN), cdiv(BATCH, TM));
        wmma_gemm_kernel<false, false><<<g, 128, 0, stream>>>(
            convb, proj_w, ybuf, proj_b, BATCH, DIMK, NFEAT, NFEAT, DIMK, DIMK,
            nullptr, nullptr, nullptr, nullptr);
    }
    // ---- 10) BN -> PReLU -> BN fused per column -> cl_x ----
    colbn_kernel<<<DIMK, 256, 0, stream>>>(ybuf, o_clx, prelu_a, BATCH, DIMK);

    // ---- 11) scores = cl_x @ ent'^T + b_out  (the 41-GFLOP / 410-MB GEMM) ----
    {
        dim3 g(cdiv(ENTN, TN), cdiv(BATCH, TM));
        wmma_gemm_kernel<true, false><<<g, 128, 0, stream>>>(
            o_clx, o_ent, o_scores, b_out, BATCH, ENTN, DIMK, DIMK, DIMK, ENTN,
            nullptr, nullptr, nullptr, nullptr);
    }
}
